// SABlock_Windows_57028575756854
// MI455X (gfx1250) — compile-verified, matrix operand order checked by reading, not run on hardware
//
#include <hip/hip_runtime.h>
#include <hip/hip_bf16.h>
#include <math.h>

typedef _Float16 half_t;
typedef __attribute__((ext_vector_type(8)))  _Float16 v8h;
typedef __attribute__((ext_vector_type(16))) _Float16 v16h;
typedef __attribute__((ext_vector_type(8)))  float    v8f;
typedef __attribute__((ext_vector_type(4)))  unsigned int uint4v;
typedef __attribute__((ext_vector_type(8)))  unsigned int uint8v;

#define DIM   384
#define NHD   12
#define HD    32
#define HH    56
#define WW    56
#define WSZ   14
#define NTOK  196     // tokens per window
#define NPAD  224     // padded to multiple of 32
#define NWIN  256     // 16 batches * 16 windows
#define MTOT  (NWIN * NTOK)   // 50176 tokens total

// ---------------- wave32 reductions ----------------
__device__ __forceinline__ float wave_sum(float v) {
#pragma unroll
  for (int o = 16; o > 0; o >>= 1) v += __shfl_xor(v, o, 32);
  return v;
}
__device__ __forceinline__ float wave_max(float v) {
#pragma unroll
  for (int o = 16; o > 0; o >>= 1) v = fmaxf(v, __shfl_xor(v, o, 32));
  return v;
}

// ---------------- TDM: 2D tile (row-major, 2-byte elems) global -> LDS ----------------
// Descriptor per CDNA5 ISA ch.8: group0 = {flags, lds_addr, gaddr_lo, gaddr_hi|type},
// group1 packs data_size/dims/strides. Issued wave-uniform; tracked by TENSORcnt.
__device__ __forceinline__ void tdm_load_2d(unsigned int lds_off,
                                            unsigned long long gaddr,
                                            unsigned int tile_d0, unsigned int tile_d1,
                                            unsigned int tens_d0, unsigned int tens_d1,
                                            unsigned int stride0) {
  uint4v g0;
  g0[0] = 1u;                                            // count=1 (valid user D#)
  g0[1] = lds_off;                                       // lds_addr (bytes)
  g0[2] = (unsigned int)(gaddr & 0xFFFFFFFFull);         // global_addr[31:0]
  g0[3] = (unsigned int)((gaddr >> 32) & 0x01FFFFFFull)  // global_addr[56:32]
          | (2u << 30);                                  // type=2 ("image")
  uint8v g1;
  g1[0] = (1u << 16);                                    // data_size=1 -> 2 bytes
  g1[1] = (tens_d0 & 0xFFFFu) << 16;                     // tensor_dim0[15:0] @ bits63:48
  g1[2] = ((tens_d0 >> 16) & 0xFFFFu) | ((tens_d1 & 0xFFFFu) << 16); // dim0 hi | dim1 lo
  g1[3] = ((tens_d1 >> 16) & 0xFFFFu) | ((tile_d0 & 0xFFFFu) << 16); // dim1 hi | tile_dim0
  g1[4] = (tile_d1 & 0xFFFFu);                           // tile_dim1 (tile_dim2=0)
  g1[5] = stride0;                                       // tensor_dim0_stride[31:0]
  g1[6] = 0u;
  g1[7] = 0u;
  asm volatile("tensor_load_to_lds %0, %1" :: "s"(g0), "s"(g1) : "memory");
}

// ---------------- f32 -> f16 weight convert ----------------
__global__ void k_f2h(const float* __restrict__ a, half_t* __restrict__ o, int n) {
  int i = blockIdx.x * 256 + threadIdx.x;
  if (i < n) o[i] = (half_t)a[i];
}

// ---------------- depthwise 3x3 conv + residual + bias, NCHW -> BHWC ----------------
__global__ void k_dwconv(const float* __restrict__ x, const float* __restrict__ cw,
                         const float* __restrict__ cb, float* __restrict__ xres) {
  int idx = blockIdx.x * 256 + threadIdx.x;       // c fastest -> coalesced BHWC store
  int c = idx % DIM;
  int t = idx / DIM;                              // raster token  b*3136 + h*56 + w
  int w = t % WW, h = (t / WW) % HH, b = t / (WW * HH);
  const float* xp = x + ((size_t)(b * DIM + c) * HH) * WW;
  float acc = 0.f;
#pragma unroll
  for (int dy = 0; dy < 3; ++dy) {
    int hh = h + dy - 1;
    if ((unsigned)hh >= (unsigned)HH) continue;
#pragma unroll
    for (int dx = 0; dx < 3; ++dx) {
      int ww2 = w + dx - 1;
      if ((unsigned)ww2 >= (unsigned)WW) continue;
      acc += cw[c * 9 + dy * 3 + dx] * xp[hh * WW + ww2];
    }
  }
  xres[(size_t)t * DIM + c] = xp[h * WW + w] + acc + cb[c];
}

// ---------------- LN1: one wave per token, write f16 in window-permuted rows ----------------
__global__ void k_ln1(const float* __restrict__ xres, const float* __restrict__ g,
                      const float* __restrict__ bt, half_t* __restrict__ yw) {
  int wv = threadIdx.x >> 5, ln = threadIdx.x & 31;
  int tok = blockIdx.x * 8 + wv;
  const float* row = xres + (size_t)tok * DIM;
  float v[12], s = 0.f;
#pragma unroll
  for (int j = 0; j < 12; ++j) { v[j] = row[ln + 32 * j]; s += v[j]; }
  float mean = wave_sum(s) * (1.f / DIM);
  float q = 0.f;
#pragma unroll
  for (int j = 0; j < 12; ++j) { float d = v[j] - mean; q += d * d; }
  float rstd = rsqrtf(wave_sum(q) * (1.f / DIM) + 1e-5f);
  int w = tok % WW, h = (tok / WW) % HH, b = tok / (WW * HH);
  int win = b * 16 + (h / WSZ) * 4 + (w / WSZ);
  int tt  = (h % WSZ) * WSZ + (w % WSZ);
  half_t* orow = yw + ((size_t)win * NTOK + tt) * DIM;
#pragma unroll
  for (int j = 0; j < 12; ++j) {
    int c = ln + 32 * j;
    orow[c] = (half_t)((v[j] - mean) * rstd * g[c] + bt[c]);
  }
}

// ---------------- WMMA GEMM, 32x32 per wave (2x2 tiles): out = A*Bw^T + bias ----------------
// MODE 0: f16 out | 1: f32 out | 2: exact-GELU, f16 out | 3: +res, scatter NCHW f32
template <int MODE>
__global__ __launch_bounds__(256) void k_gemm(
    const half_t* __restrict__ A, const half_t* __restrict__ Bw,
    const float* __restrict__ bias, const float* __restrict__ res,
    void* __restrict__ out, int M, int N, int K) {
  int wv = threadIdx.x >> 5, ln = threadIdx.x & 31;
  int mt0 = (blockIdx.y * 8 + wv) * 2;
  int nt0 = blockIdx.x * 2;
  int l16 = ln & 15, hl = ln >> 4;            // K-offset pattern: {0,8} then +16
  const half_t* ap[2];
  const half_t* bp[2];
  ap[0] = A  + (size_t)(mt0 * 16 + l16) * K + hl * 8;
  ap[1] = ap[0] + (size_t)16 * K;
  bp[0] = Bw + (size_t)(nt0 * 16 + l16) * K + hl * 8;
  bp[1] = bp[0] + (size_t)16 * K;
  v8f acc[2][2] = {};
  for (int k0 = 0; k0 < K; k0 += 32) {
    union { v16h v; v8h h[2]; } af[2], bf[2];
#pragma unroll
    for (int t = 0; t < 2; ++t) {
      af[t].h[0] = *(const v8h*)(ap[t]);
      af[t].h[1] = *(const v8h*)(ap[t] + 16);
      bf[t].h[0] = *(const v8h*)(bp[t]);
      bf[t].h[1] = *(const v8h*)(bp[t] + 16);
      ap[t] += 32; bp[t] += 32;
      __builtin_prefetch(ap[t] + 96, 0, 0);   // global_prefetch_b8
      __builtin_prefetch(bp[t] + 96, 0, 0);
    }
#pragma unroll
    for (int ti = 0; ti < 2; ++ti)
#pragma unroll
      for (int tj = 0; tj < 2; ++tj)
        acc[ti][tj] = __builtin_amdgcn_wmma_f32_16x16x32_f16(
            false, af[ti].v, false, bf[tj].v, (short)0, acc[ti][tj], false, false);
  }
#pragma unroll
  for (int ti = 0; ti < 2; ++ti) {
#pragma unroll
    for (int tj = 0; tj < 2; ++tj) {
      int n = (nt0 + tj) * 16 + l16;
      float bn = bias[n];
#pragma unroll
      for (int i = 0; i < 8; ++i) {
        int m = (mt0 + ti) * 16 + i + 8 * hl;   // C/D layout: lanes16-31 -> M+8
        float v = acc[ti][tj][i] + bn;
        if (MODE == 0) {
          ((half_t*)out)[(size_t)m * N + n] = (half_t)v;
        } else if (MODE == 1) {
          ((float*)out)[(size_t)m * N + n] = v;
        } else if (MODE == 2) {
          float gl = 0.5f * v * (1.f + erff(v * 0.70710678118654752f));
          ((half_t*)out)[(size_t)m * N + n] = (half_t)gl;
        } else {
          v += res[(size_t)m * N + n];
          int w = m % WW, h = (m / WW) % HH, b = m / (WW * HH);
          ((float*)out)[(((size_t)b * DIM + n) * HH + h) * WW + w] = v;
        }
      }
    }
  }
}

// ---------------- windowed attention: one block per (window, head) ----------------
__global__ __launch_bounds__(256) void k_attn(const half_t* __restrict__ qkv,
                                              half_t* __restrict__ oh) {
  __shared__ __attribute__((aligned(16))) half_t qs[NPAD * HD];     // 14336 B
  __shared__ __attribute__((aligned(16))) half_t ks_[NPAD * HD];    // 14336 B
  __shared__ __attribute__((aligned(16))) half_t vT[HD * NPAD];     // 14336 B (transposed)
  __shared__ __attribute__((aligned(16))) float  sb[16 * NPAD];     // 14336 B
  __shared__ __attribute__((aligned(16))) half_t pb[16 * NPAD];     //  7168 B
  __shared__ float inv[16];
  int tid = threadIdx.x;
  int wv = tid >> 5, ln = tid & 31;
  int head = blockIdx.x, win = blockIdx.y;
  const half_t* base = qkv + (size_t)win * NTOK * (3 * DIM) + head * HD;

  // stage V manually (transposed for K-contiguous B reads); zero qs/ks pad rows
  for (int idx = tid; idx < NPAD * 4; idx += 256) {
    int row = idx >> 2, ch = (idx & 3) * 8;
    v8h v8 = {};
    if (row < NTOK) {
      v8 = *(const v8h*)(base + (size_t)row * (3 * DIM) + 2 * DIM + ch);
    } else {
      *(v8h*)(qs  + row * HD + ch) = v8;
      *(v8h*)(ks_ + row * HD + ch) = v8;
    }
#pragma unroll
    for (int j = 0; j < 8; ++j) vT[(ch + j) * NPAD + row] = v8[j];
  }
  // q,k tiles via Tensor Data Mover (wave 0 issues; TENSORcnt waited before barrier)
  if (tid < 32) {
    tdm_load_2d((unsigned int)(uintptr_t)qs,
                (unsigned long long)(uintptr_t)base,
                HD, NTOK, HD, NTOK, 3 * DIM);
    tdm_load_2d((unsigned int)(uintptr_t)ks_,
                (unsigned long long)(uintptr_t)(base + DIM),
                HD, NTOK, HD, NTOK, 3 * DIM);
    __builtin_amdgcn_s_wait_tensorcnt(0);
  }
  __syncthreads();

  int l16 = ln & 15, hl = ln >> 4;
  const float scale = 0.17677669529663687f;   // 1/sqrt(32)

  for (int mtile = 0; mtile < 13; ++mtile) {   // 13 row-blocks cover 196 (+pad) rows
    // ---- S block = q_block * K^T  (13 tiles of 16 cols, K=32 -> 1 WMMA each)
    for (int ntl = wv; ntl < 13; ntl += 8) {
      union { v16h v; v8h h[2]; } a, b;
      const half_t* qp = qs  + (mtile * 16 + l16) * HD + hl * 8;
      a.h[0] = *(const v8h*)(qp);
      a.h[1] = *(const v8h*)(qp + 16);
      const half_t* kp = ks_ + (ntl * 16 + l16) * HD + hl * 8;
      b.h[0] = *(const v8h*)(kp);
      b.h[1] = *(const v8h*)(kp + 16);
      v8f acc = {};
      acc = __builtin_amdgcn_wmma_f32_16x16x32_f16(false, a.v, false, b.v,
                                                   (short)0, acc, false, false);
#pragma unroll
      for (int i = 0; i < 8; ++i)
        sb[(i + 8 * hl) * NPAD + ntl * 16 + l16] = acc[i] * scale;
    }
    __syncthreads();
    // ---- softmax over valid 196 cols; P kept f16, 1/sum deferred to PV epilogue
    for (int r = wv; r < 16; r += 8) {
      float vals[7];
#pragma unroll
      for (int j = 0; j < 7; ++j) {
        int col = ln + 32 * j;
        vals[j] = (col < NTOK) ? sb[r * NPAD + col] : -1e30f;
      }
      float mx = vals[0];
#pragma unroll
      for (int j = 1; j < 7; ++j) mx = fmaxf(mx, vals[j]);
      mx = wave_max(mx);
      float s = 0.f;
#pragma unroll
      for (int j = 0; j < 7; ++j) {
        float e = __expf(vals[j] - mx);       // padded cols -> 0
        s += e;
        pb[r * NPAD + ln + 32 * j] = (half_t)e;
      }
      s = wave_sum(s);
      if (ln == 0) inv[r] = (s > 0.f) ? (1.f / s) : 0.f;
    }
    __syncthreads();
    // ---- O block = P_block(16x224) * V(224x32): 2 n-tiles x 7 K-steps
    if (wv < 2) {
      int ntl = wv;
      v8f acc = {};
#pragma unroll
      for (int kk = 0; kk < 7; ++kk) {
        union { v16h v; v8h h[2]; } a, b;
        const half_t* pp = pb + l16 * NPAD + kk * 32 + hl * 8;
        a.h[0] = *(const v8h*)(pp);
        a.h[1] = *(const v8h*)(pp + 16);
        const half_t* vp = vT + (ntl * 16 + l16) * NPAD + kk * 32 + hl * 8;
        b.h[0] = *(const v8h*)(vp);
        b.h[1] = *(const v8h*)(vp + 16);
        acc = __builtin_amdgcn_wmma_f32_16x16x32_f16(false, a.v, false, b.v,
                                                     (short)0, acc, false, false);
      }
#pragma unroll
      for (int i = 0; i < 8; ++i) {
        int mloc = i + 8 * hl;
        int m = mtile * 16 + mloc;
        if (m < NTOK) {
          float o = acc[i] * inv[mloc];
          oh[((size_t)win * NTOK + m) * DIM + head * HD + ntl * 16 + l16] = (half_t)o;
        }
      }
    }
    __syncthreads();
  }
}

// ---------------- proj residual (window->raster) + LN2; x updated in place ----------------
__global__ void k_res_ln2(const float* __restrict__ po, float* __restrict__ xres,
                          const float* __restrict__ g, const float* __restrict__ bt,
                          half_t* __restrict__ z) {
  int wv = threadIdx.x >> 5, ln = threadIdx.x & 31;
  int tok = blockIdx.x * 8 + wv;
  int w = tok % WW, h = (tok / WW) % HH, b = tok / (WW * HH);
  int win = b * 16 + (h / WSZ) * 4 + (w / WSZ);
  int tt  = (h % WSZ) * WSZ + (w % WSZ);
  const float* prow = po + ((size_t)win * NTOK + tt) * DIM;
  float* xrow = xres + (size_t)tok * DIM;
  float v[12], s = 0.f;
#pragma unroll
  for (int j = 0; j < 12; ++j) {
    int c = ln + 32 * j;
    v[j] = xrow[c] + prow[c];
    s += v[j];
  }
  float mean = wave_sum(s) * (1.f / DIM);
  float q = 0.f;
#pragma unroll
  for (int j = 0; j < 12; ++j) { float d = v[j] - mean; q += d * d; }
  float rstd = rsqrtf(wave_sum(q) * (1.f / DIM) + 1e-5f);
#pragma unroll
  for (int j = 0; j < 12; ++j) {
    int c = ln + 32 * j;
    xrow[c] = v[j];
    z[(size_t)tok * DIM + c] = (half_t)((v[j] - mean) * rstd * g[c] + bt[c]);
  }
}

extern "C" void kernel_launch(void* const* d_in, const int* in_sizes, int n_in,
                              void* d_out, int out_size, void* d_ws, size_t ws_size,
                              hipStream_t stream) {
  (void)in_sizes; (void)n_in; (void)out_size; (void)ws_size;
  const float* x      = (const float*)d_in[0];
  const float* conv_w = (const float*)d_in[1];
  const float* conv_b = (const float*)d_in[2];
  const float* ln1_g  = (const float*)d_in[3];
  const float* ln1_b  = (const float*)d_in[4];
  const float* qkv_w  = (const float*)d_in[5];
  const float* qkv_b  = (const float*)d_in[6];
  const float* proj_w = (const float*)d_in[7];
  const float* proj_b = (const float*)d_in[8];
  const float* ln2_g  = (const float*)d_in[9];
  const float* ln2_b  = (const float*)d_in[10];
  const float* fc1_w  = (const float*)d_in[11];
  const float* fc1_b  = (const float*)d_in[12];
  const float* fc2_w  = (const float*)d_in[13];
  const float* fc2_b  = (const float*)d_in[14];

  char* ws = (char*)d_ws;
  size_t off = 0;
  auto wsalloc = [&](size_t bytes) -> char* {
    char* p = ws + off;
    off = (off + bytes + 255) & ~(size_t)255;
    return p;
  };
  float*  xres  = (float*) wsalloc((size_t)MTOT * DIM * 4);       // 77 MB, lives whole pass
  half_t* act16 = (half_t*)wsalloc((size_t)MTOT * DIM * 2);       // 38 MB: ln1 -> attnO -> ln2
  char*   big   = wsalloc((size_t)MTOT * 4 * DIM * 2);            // 154 MB: qkv -> projO -> fc1h
  half_t* wq = (half_t*)wsalloc((size_t)3 * DIM * DIM * 2);
  half_t* wp = (half_t*)wsalloc((size_t)DIM * DIM * 2);
  half_t* w1 = (half_t*)wsalloc((size_t)4 * DIM * DIM * 2);
  half_t* w2 = (half_t*)wsalloc((size_t)4 * DIM * DIM * 2);

  // weights -> f16 (tiny; stays in L2)
  k_f2h<<<(3 * DIM * DIM + 255) / 256, 256, 0, stream>>>(qkv_w, wq, 3 * DIM * DIM);
  k_f2h<<<(DIM * DIM + 255) / 256, 256, 0, stream>>>(proj_w, wp, DIM * DIM);
  k_f2h<<<(4 * DIM * DIM + 255) / 256, 256, 0, stream>>>(fc1_w, w1, 4 * DIM * DIM);
  k_f2h<<<(4 * DIM * DIM + 255) / 256, 256, 0, stream>>>(fc2_w, w2, 4 * DIM * DIM);

  // conv pos-embed residual (NCHW -> BHWC f32)
  k_dwconv<<<(MTOT * DIM) / 256, 256, 0, stream>>>(x, conv_w, conv_b, xres);
  // LN1 -> f16, window-permuted token rows
  k_ln1<<<MTOT / 8, 256, 0, stream>>>(xres, ln1_g, ln1_b, act16);
  // qkv = y @ qkv_w^T + b  -> f16 [MTOT x 1152]
  k_gemm<0><<<dim3(3 * DIM / 32, MTOT / 256), 256, 0, stream>>>(
      act16, wq, qkv_b, nullptr, big, MTOT, 3 * DIM, DIM);
  // windowed attention -> f16 [MTOT x 384] (window order)
  k_attn<<<dim3(NHD, NWIN), 256, 0, stream>>>((const half_t*)big, act16);
  // proj -> f32 [MTOT x 384] (window order)
  k_gemm<1><<<dim3(DIM / 32, MTOT / 256), 256, 0, stream>>>(
      act16, wp, proj_b, nullptr, big, MTOT, DIM, DIM);
  // residual (window reverse) + LN2 -> f16 raster rows; xres updated in place
  k_res_ln2<<<MTOT / 8, 256, 0, stream>>>(
      (const float*)big, xres, ln2_g, ln2_b, act16);
  // fc1 + exact GELU -> f16 [MTOT x 1536]
  k_gemm<2><<<dim3(4 * DIM / 32, MTOT / 256), 256, 0, stream>>>(
      act16, w1, fc1_b, nullptr, big, MTOT, 4 * DIM, DIM);
  // fc2 + residual, scatter to NCHW f32 output
  k_gemm<3><<<dim3(DIM / 32, MTOT / 256), 256, 0, stream>>>(
      (const half_t*)big, w2, fc2_b, xres, d_out, MTOT, DIM, 4 * DIM);
}
